// GAT_10282151706737
// MI455X (gfx1250) — matrix operand order repaired and checked by reading, not compiled, not run on hardware
//
#include <hip/hip_runtime.h>
#include <math.h>

// ---------------- problem constants (from reference) ----------------
#define GN 50000      // nodes
#define GE 120000     // edges (before self loops)
#define GD 768        // input dim
#define GH 3          // heads
#define GC 768        // channels per head
#define GHC (GH * GC) // 2304
#define NEG_SLOPE 0.2f
#define GEPS 1e-16f

typedef unsigned short u16_t;
typedef unsigned int   u32_t;

typedef __attribute__((ext_vector_type(16))) __bf16 v16bf;
typedef __attribute__((ext_vector_type(8)))  __bf16 v8bf;
typedef __attribute__((ext_vector_type(8)))  float  v8f;

// CDNA5 async global->LDS staging (ASYNCcnt-tracked), if the toolchain has it.
// Probe round 3: param 0 is 'int __vector(4) __device__ *' (AS1, non-const int4*).
#if __has_builtin(__builtin_amdgcn_global_load_async_to_lds_b128) && \
    __has_builtin(__builtin_amdgcn_s_wait_asynccnt)
#define USE_ASYNC_LDS 1
typedef int vi4_t __attribute__((ext_vector_type(4)));
typedef vi4_t __attribute__((address_space(1)))* glb_vi4_p;
typedef vi4_t __attribute__((address_space(3)))* lds_vi4_p;
#define AS_GLOBAL(p) ((glb_vi4_p)(p))
#define AS_LOCAL(p)  ((lds_vi4_p)(p))
#else
#define USE_ASYNC_LDS 0
#endif

// ---------------- small helpers ----------------
__device__ __forceinline__ u16_t f2bf(float f) {
    u32_t u = __float_as_uint(f);
    u32_t r = (u + 0x7FFFu + ((u >> 16) & 1u)) >> 16;   // RNE
    return (u16_t)r;
}

__device__ __forceinline__ v8bf as_v8bf(uint4 u) {
    union { uint4 u; v8bf b; } x; x.u = u; return x.b;
}

__device__ __forceinline__ void atomicMaxFloat(float* addr, float val) {
    // works with init = -inf; classic sign-split trick
    if (val >= 0.0f) atomicMax((int*)addr, __float_as_int(val));
    else             atomicMin((u32_t*)addr, __float_as_uint(val));
}

__device__ __forceinline__ float lrelu(float x) {
    return x > 0.0f ? x : NEG_SLOPE * x;
}

// ---------------- f32 -> bf16 conversion ----------------
__global__ void k_f32_to_bf16(const float* __restrict__ in,
                              u16_t* __restrict__ out, size_t n) {
    size_t i = (size_t)blockIdx.x * blockDim.x + threadIdx.x;
    size_t stride = (size_t)gridDim.x * blockDim.x;
    for (; i < n; i += stride) out[i] = f2bf(in[i]);
}

// ---------------- W [K,Nout] f32  ->  Wt [Nout,K] bf16 (LDS-tiled) ----------------
__global__ __launch_bounds__(256)
void k_transpose_w_bf16(const float* __restrict__ W,  // [K, Nout]
                        u16_t* __restrict__ Wt,       // [Nout, K]
                        int K, int Nout) {
    __shared__ u16_t tile[32][33];
    int n0 = blockIdx.x * 32;
    int k0 = blockIdx.y * 32;
    int tx = threadIdx.x & 31;
    int ty = threadIdx.x >> 5;     // 0..7
#pragma unroll
    for (int dy = 0; dy < 32; dy += 8) {
        int k = k0 + ty + dy, n = n0 + tx;
        tile[ty + dy][tx] = f2bf(W[(size_t)k * Nout + n]);
    }
    __syncthreads();
#pragma unroll
    for (int dy = 0; dy < 32; dy += 8) {
        int n = n0 + ty + dy, k = k0 + tx;
        Wt[(size_t)n * K + k] = tile[tx][ty + dy];
    }
}

// ---------------- fill f32 ----------------
__global__ void k_fill_f32(float* __restrict__ p, float v, int n) {
    int i = blockIdx.x * blockDim.x + threadIdx.x;
    if (i < n) p[i] = v;
}

// ---------------- bf16 WMMA GEMM:  C[M,Nout] = A[M,K] * Bt[Nout,K]^T ----------------
// block tile 128x128, BK=32, 256 threads (8 waves as 2x4), wave tile 64x32.
// Double-buffered LDS (ping-pong), 1 barrier per K-step, prefetch 2 steps ahead.
#define LDT 40   // padded LDS row stride in bf16 elems (80B, 16B-aligned)

__global__ __launch_bounds__(256)
void k_gemm_bf16(const u16_t* __restrict__ A,   // [M,K]  bf16 (row major, K contig)
                 const u16_t* __restrict__ Bt,  // [Nout,K] bf16 (pre-transposed W)
                 float* __restrict__ C,         // [M,Nout] f32
                 int M, int K, int Nout) {
    __shared__ u16_t sA[2][128 * LDT];   // [buf][m][k]
    __shared__ u16_t sB[2][128 * LDT];   // [buf][n][k]

    const int bm   = blockIdx.x;
    const int bn   = blockIdx.y;
    const int tid  = threadIdx.x;
    const int lane = tid & 31;
    const int wave = tid >> 5;
    const int wm   = wave >> 2;   // 0..1
    const int wn   = wave & 3;    // 0..3
    const int half = lane >> 4;   // 0..1
    const int r    = lane & 15;   // 0..15

    // per-thread staging addresses: 16 bf16 (2 x 16B) per tile per matrix.
    // OOB A rows are clamped to row M-1: sA row r only ever feeds output row
    // bm*128+r, which is store-guarded, so stale data there is harmless.
    const int row2 = tid >> 1;          // 0..127
    const int col2 = (tid & 1) * 16;    // 0 or 16
    const int grA  = bm * 128 + row2;
    const bool okA = grA < M;
    const u16_t* pA = A  + (size_t)(okA ? grA : (M - 1)) * K + col2;
    const u16_t* pB = Bt + (size_t)(bn * 128 + row2)     * K + col2;
    u16_t* dA = &sA[0][row2 * LDT + col2];
    u16_t* dB = &sB[0][row2 * LDT + col2];
    const int bufStep = 128 * LDT;       // elems between ping-pong buffers

    v8f acc[4][2];
    const v8f vzero = {0.f,0.f,0.f,0.f,0.f,0.f,0.f,0.f};
#pragma unroll
    for (int i = 0; i < 4; ++i)
#pragma unroll
        for (int j = 0; j < 2; ++j) acc[i][j] = vzero;

    const int iters = K / 32;

    // ---- preload tile 0 into buffer 0 ----
#if USE_ASYNC_LDS
    __builtin_amdgcn_global_load_async_to_lds_b128(AS_GLOBAL(pA),     AS_LOCAL(dA),     0, 0);
    __builtin_amdgcn_global_load_async_to_lds_b128(AS_GLOBAL(pA + 8), AS_LOCAL(dA + 8), 0, 0);
    __builtin_amdgcn_global_load_async_to_lds_b128(AS_GLOBAL(pB),     AS_LOCAL(dB),     0, 0);
    __builtin_amdgcn_global_load_async_to_lds_b128(AS_GLOBAL(pB + 8), AS_LOCAL(dB + 8), 0, 0);
    __builtin_amdgcn_s_wait_asynccnt(0);
#else
    {
        uint4 a0 = *(const uint4*)(pA);
        uint4 a1 = *(const uint4*)(pA + 8);
        uint4 b0 = *(const uint4*)(pB);
        uint4 b1 = *(const uint4*)(pB + 8);
        *(uint4*)(dA)     = a0;
        *(uint4*)(dA + 8) = a1;
        *(uint4*)(dB)     = b0;
        *(uint4*)(dB + 8) = b1;
    }
#endif
    __syncthreads();

    for (int it = 0; it < iters; ++it) {
        const int cur = it & 1;
        const int nxt = cur ^ 1;
        const bool have = (it + 1) < iters;

        // L2 prefetch two steps ahead (global_prefetch_b8)
        if (it + 2 < iters) {
            __builtin_prefetch(pA + (size_t)(it + 2) * 32, 0, 0);
            __builtin_prefetch(pB + (size_t)(it + 2) * 32, 0, 0);
        }

#if USE_ASYNC_LDS
        // issue async global->LDS copies for the next tile; they land in the
        // other buffer while WMMAs consume the current one.
        if (have) {
            const u16_t* qA = pA + (size_t)(it + 1) * 32;
            const u16_t* qB = pB + (size_t)(it + 1) * 32;
            u16_t* tA = dA + nxt * bufStep;
            u16_t* tB = dB + nxt * bufStep;
            __builtin_amdgcn_global_load_async_to_lds_b128(AS_GLOBAL(qA),     AS_LOCAL(tA),     0, 0);
            __builtin_amdgcn_global_load_async_to_lds_b128(AS_GLOBAL(qA + 8), AS_LOCAL(tA + 8), 0, 0);
            __builtin_amdgcn_global_load_async_to_lds_b128(AS_GLOBAL(qB),     AS_LOCAL(tB),     0, 0);
            __builtin_amdgcn_global_load_async_to_lds_b128(AS_GLOBAL(qB + 8), AS_LOCAL(tB + 8), 0, 0);
        }
#else
        uint4 a0, a1, b0, b1;
        if (have) {
            const u16_t* qA = pA + (size_t)(it + 1) * 32;
            const u16_t* qB = pB + (size_t)(it + 1) * 32;
            a0 = *(const uint4*)(qA);
            a1 = *(const uint4*)(qA + 8);
            b0 = *(const uint4*)(qB);
            b1 = *(const uint4*)(qB + 8);
        }
#endif

        // ---- B fragments (elems 0..7: K = half*8 + 0..7; elems 8..15: K = 16 + half*8 + 0..7) ----
        v16bf bfrag[2];
#pragma unroll
        for (int j = 0; j < 2; ++j) {
            int col = wn * 32 + j * 16 + r;
            const u16_t* p = &sB[cur][col * LDT + half * 8];
            v8bf lo = as_v8bf(*(const uint4*)(p));
            v8bf hi = as_v8bf(*(const uint4*)(p + 16));
            bfrag[j] = __builtin_shufflevector(lo, hi,
                0,1,2,3,4,5,6,7,8,9,10,11,12,13,14,15);
        }
        // ---- A fragments + 8 WMMAs ----
#pragma unroll
        for (int i = 0; i < 4; ++i) {
            int row = wm * 64 + i * 16 + r;
            // elems 0..7 : K = half*8 + 0..7 ; elems 8..15 : K = 16 + half*8 + 0..7
            v8bf lo = as_v8bf(*(const uint4*)&sA[cur][row * LDT + half * 8]);
            v8bf hi = as_v8bf(*(const uint4*)&sA[cur][row * LDT + 16 + half * 8]);
            v16bf afrag = __builtin_shufflevector(lo, hi,
                0,1,2,3,4,5,6,7,8,9,10,11,12,13,14,15);
#pragma unroll
            for (int j = 0; j < 2; ++j)
                acc[i][j] = __builtin_amdgcn_wmma_f32_16x16x32_bf16(
                    false, afrag, false, bfrag[j],
                    (short)0, acc[i][j], false, false);
        }

#if USE_ASYNC_LDS
        if (have) __builtin_amdgcn_s_wait_asynccnt(0);
#else
        if (have) {
            u16_t* tA = dA + nxt * bufStep;
            u16_t* tB = dB + nxt * bufStep;
            *(uint4*)(tA)     = a0;
            *(uint4*)(tA + 8) = a1;
            *(uint4*)(tB)     = b0;
            *(uint4*)(tB + 8) = b1;
        }
#endif
        __syncthreads();
    }

    // ---- store: D layout  M = half*8 + v,  N = lane&15 ----
    const int baseRow = bm * 128 + wm * 64 + half * 8;
    const int baseCol = bn * 128 + wn * 32 + r;
    if (bm * 128 + 128 <= M) {
        // fast path: whole block in range, no per-element guards
#pragma unroll
        for (int i = 0; i < 4; ++i) {
#pragma unroll
            for (int j = 0; j < 2; ++j) {
                float* cp = C + (size_t)(baseRow + i * 16) * Nout + baseCol + j * 16;
#pragma unroll
                for (int v = 0; v < 8; ++v)
                    cp[(size_t)v * Nout] = acc[i][j][v];
            }
        }
    } else {
#pragma unroll
        for (int i = 0; i < 4; ++i) {
#pragma unroll
            for (int j = 0; j < 2; ++j) {
#pragma unroll
                for (int v = 0; v < 8; ++v) {
                    int grow = baseRow + i * 16 + v;
                    if (grow < M)
                        C[(size_t)grow * Nout + baseCol + j * 16] = acc[i][j][v];
                }
            }
        }
    }
}

// ---------------- attention logits: one wave per (node, head) ----------------
__global__ __launch_bounds__(256)
void k_att_logits(const float* __restrict__ h,      // [N, H*C]
                  const float* __restrict__ a_src,  // [H, C]
                  const float* __restrict__ a_dst,  // [H, C]
                  float* __restrict__ al_s,         // [N*H]
                  float* __restrict__ al_d,         // [N*H]
                  int NH) {
    int w = blockIdx.x * 8 + (threadIdx.x >> 5);
    if (w >= NH) return;                       // uniform per wave
    int lane = threadIdx.x & 31;
    int n  = w / GH;
    int hh = w % GH;
    const float* hp = h + (size_t)n * GHC + hh * GC;
    const float* as = a_src + hh * GC;
    const float* ad = a_dst + hh * GC;
    float ss = 0.f, sd = 0.f;
    for (int c = lane; c < GC; c += 32) {
        float v = hp[c];
        ss += v * as[c];
        sd += v * ad[c];
    }
#pragma unroll
    for (int off = 16; off > 0; off >>= 1) {
        ss += __shfl_down(ss, off, 32);
        sd += __shfl_down(sd, off, 32);
    }
    if (lane == 0) { al_s[w] = ss; al_d[w] = sd; }
}

// ---------------- segment max over edges (incl. self loops) ----------------
__global__ void k_edge_max(const long long* __restrict__ ei,
                           const float* __restrict__ al_s,
                           const float* __restrict__ al_d,
                           float* __restrict__ m) {
    int e = blockIdx.x * blockDim.x + threadIdx.x;
    int tot = GE + GN;
    if (e >= tot) return;
    int src, dst;
    if (e < GE) { src = (int)ei[e]; dst = (int)ei[GE + e]; }
    else        { src = dst = e - GE; }
#pragma unroll
    for (int hh = 0; hh < GH; ++hh) {
        float x = lrelu(al_s[src * GH + hh] + al_d[dst * GH + hh]);
        atomicMaxFloat(&m[dst * GH + hh], x);
    }
}

// ---------------- segment sum of exp(e - m) ----------------
__global__ void k_edge_expsum(const long long* __restrict__ ei,
                              const float* __restrict__ al_s,
                              const float* __restrict__ al_d,
                              const float* __restrict__ m,
                              float* __restrict__ s) {
    int e = blockIdx.x * blockDim.x + threadIdx.x;
    int tot = GE + GN;
    if (e >= tot) return;
    int src, dst;
    if (e < GE) { src = (int)ei[e]; dst = (int)ei[GE + e]; }
    else        { src = dst = e - GE; }
#pragma unroll
    for (int hh = 0; hh < GH; ++hh) {
        float x = lrelu(al_s[src * GH + hh] + al_d[dst * GH + hh]);
        atomicAdd(&s[dst * GH + hh], expf(x - m[dst * GH + hh]));
    }
}

// ---------------- aggregation fused with head-mean ----------------
// out[dst, c] += (1/H) * sum_h alpha[e,h] * h[src, h, c]
__global__ __launch_bounds__(256)
void k_edge_aggregate(const long long* __restrict__ ei,
                      const float* __restrict__ h,     // [N, H*C]
                      const float* __restrict__ al_s,
                      const float* __restrict__ al_d,
                      const float* __restrict__ m,
                      const float* __restrict__ s,
                      float* __restrict__ out) {       // [N, C]
    int e = blockIdx.x;
    int src, dst;
    if (e < GE) { src = (int)ei[e]; dst = (int)ei[GE + e]; }
    else        { src = dst = e - GE; }
    float alpha[GH];
#pragma unroll
    for (int hh = 0; hh < GH; ++hh) {
        float x = lrelu(al_s[src * GH + hh] + al_d[dst * GH + hh]);
        alpha[hh] = expf(x - m[dst * GH + hh]) /
                    (s[dst * GH + hh] + GEPS) * (1.0f / (float)GH);
    }
    const float* hs = h + (size_t)src * GHC;
    float* op = out + (size_t)dst * GC;
    for (int c = threadIdx.x; c < GC; c += 256) {
        float acc = alpha[0] * hs[c]
                  + alpha[1] * hs[GC + c]
                  + alpha[2] * hs[2 * GC + c];
        atomicAdd(&op[c], acc);
    }
}

// ---------------- finalize layer 1: +bias, relu, quantize to bf16 ----------------
__global__ void k_finalize_relu_bf16(const float* __restrict__ in,  // [N, C]
                                     const float* __restrict__ bias,
                                     u16_t* __restrict__ out_bf,    // [N, C]
                                     size_t total) {
    size_t i = (size_t)blockIdx.x * blockDim.x + threadIdx.x;
    size_t stride = (size_t)gridDim.x * blockDim.x;
    for (; i < total; i += stride) {
        int c = (int)(i % GC);
        float v = in[i] + bias[c];
        v = v > 0.f ? v : 0.f;
        out_bf[i] = f2bf(v);
    }
}

// ---------------- finalize layer 2: +bias in place ----------------
__global__ void k_add_bias(float* __restrict__ io, const float* __restrict__ bias,
                           size_t total) {
    size_t i = (size_t)blockIdx.x * blockDim.x + threadIdx.x;
    size_t stride = (size_t)gridDim.x * blockDim.x;
    for (; i < total; i += stride) io[i] += bias[(int)(i % GC)];
}

// =====================================================================
extern "C" void kernel_launch(void* const* d_in, const int* in_sizes, int n_in,
                              void* d_out, int out_size, void* d_ws, size_t ws_size,
                              hipStream_t stream) {
    const float*     x      = (const float*)d_in[0];
    const long long* ei     = (const long long*)d_in[1];   // int64 [2, E]
    const float*     W1     = (const float*)d_in[2];
    const float*     a_src1 = (const float*)d_in[3];
    const float*     a_dst1 = (const float*)d_in[4];
    const float*     b1     = (const float*)d_in[5];
    const float*     W2     = (const float*)d_in[6];
    const float*     a_src2 = (const float*)d_in[7];
    const float*     a_dst2 = (const float*)d_in[8];
    const float*     b2     = (const float*)d_in[9];
    float* out = (float*)d_out;

    // ---- carve workspace ----
    char* wp = (char*)d_ws;
    auto carve = [&](size_t bytes) -> void* {
        void* p = (void*)wp;
        wp += (bytes + 255) & ~(size_t)255;
        return p;
    };
    float* hbuf  = (float*)carve((size_t)GN * GHC * 4);   // 460.8 MB
    u16_t* xb    = (u16_t*)carve((size_t)GN * GD  * 2);   //  76.8 MB
    u16_t* w1t   = (u16_t*)carve((size_t)GD * GHC * 2);   // Wt: [Nout,K] bf16
    u16_t* w2t   = (u16_t*)carve((size_t)GD * GHC * 2);
    float* als   = (float*)carve((size_t)GN * GH * 4);
    float* ald   = (float*)carve((size_t)GN * GH * 4);
    float* mbuf  = (float*)carve((size_t)GN * GH * 4);
    float* sbuf  = (float*)carve((size_t)GN * GH * 4);
    float* out1  = (float*)carve((size_t)GN * GC * 4);    // 153.6 MB
    (void)ws_size; (void)in_sizes; (void)n_in; (void)out_size;

    const int ETOT   = GE + GN;
    const int NH     = GN * GH;
    const dim3 gemmGrid((GN + 127) / 128, GHC / 128);     // 391 x 18
    const dim3 trGrid(GHC / 32, GD / 32);                 // 72 x 24

    // ---- weight transpose+convert, input conversion to bf16 ----
    k_transpose_w_bf16<<<trGrid, 256, 0, stream>>>(W1, w1t, GD, GHC);
    k_transpose_w_bf16<<<trGrid, 256, 0, stream>>>(W2, w2t, GD, GHC);
    k_f32_to_bf16<<<4096, 256, 0, stream>>>(x, xb, (size_t)GN * GD);

    // ================= layer 1 =================
    k_gemm_bf16<<<gemmGrid, 256, 0, stream>>>(xb, w1t, hbuf, GN, GD, GHC);

    k_att_logits<<<(NH + 7) / 8, 256, 0, stream>>>(hbuf, a_src1, a_dst1, als, ald, NH);

    k_fill_f32<<<(NH + 255) / 256, 256, 0, stream>>>(mbuf, -INFINITY, NH);
    (void)hipMemsetAsync(sbuf, 0, (size_t)NH * 4, stream);
    (void)hipMemsetAsync(out1, 0, (size_t)GN * GC * 4, stream);

    k_edge_max   <<<(ETOT + 255) / 256, 256, 0, stream>>>(ei, als, ald, mbuf);
    k_edge_expsum<<<(ETOT + 255) / 256, 256, 0, stream>>>(ei, als, ald, mbuf, sbuf);
    k_edge_aggregate<<<ETOT, 256, 0, stream>>>(ei, hbuf, als, ald, mbuf, sbuf, out1);

    k_finalize_relu_bf16<<<4096, 256, 0, stream>>>(out1, b1, xb, (size_t)GN * GC);

    // ================= layer 2 =================
    k_gemm_bf16<<<gemmGrid, 256, 0, stream>>>(xb, w2t, hbuf, GN, GD, GHC);

    k_att_logits<<<(NH + 7) / 8, 256, 0, stream>>>(hbuf, a_src2, a_dst2, als, ald, NH);

    k_fill_f32<<<(NH + 255) / 256, 256, 0, stream>>>(mbuf, -INFINITY, NH);
    (void)hipMemsetAsync(sbuf, 0, (size_t)NH * 4, stream);
    (void)hipMemsetAsync(out, 0, (size_t)GN * GC * 4, stream);

    k_edge_max   <<<(ETOT + 255) / 256, 256, 0, stream>>>(ei, als, ald, mbuf);
    k_edge_expsum<<<(ETOT + 255) / 256, 256, 0, stream>>>(ei, als, ald, mbuf, sbuf);
    k_edge_aggregate<<<ETOT, 256, 0, stream>>>(ei, hbuf, als, ald, mbuf, sbuf, out);

    k_add_bias<<<4096, 256, 0, stream>>>(out, b2, (size_t)GN * GC);
}
